// CrossAttention_24902220382250
// MI455X (gfx1250) — matrix operand order repaired and checked by reading, not compiled, not run on hardware
//
#include <hip/hip_runtime.h>
#include <hip/hip_bf16.h>
#include <math.h>

// ---------------------------------------------------------------------------
// CrossAttention on MI455X (gfx1250, wave32, WMMA).
// v_wmma_f32_16x16x32_bf16 for all GEMMs, double-buffered LDS pipeline,
// global_load_async_to_lds_b128 staging for bf16 operands (ASYNCcnt path),
// native bf16 converts for fp32 operands.
// ---------------------------------------------------------------------------

typedef unsigned short u16;
typedef __attribute__((ext_vector_type(16))) __bf16 v16bf;
typedef __attribute__((ext_vector_type(8)))  float  v8f;

#define BM 128
#define BN 128
#define BK 64
#define NTHREADS 256
#define LDS_STRIDE 72   // 64 K-elems + 8 pad (u16); 144B rows -> conflict-free b128 reads

enum OutMode { OUT_BF16_N = 0, OUT_BF16_T = 1, OUT_F32_SCORES = 2, OUT_F32_PLAIN = 3 };

// Native RNE converts (hardware v_cvt on gfx1250).
__device__ __forceinline__ unsigned pack2_bf16(float a, float b) {
  union { __bf16 h[2]; unsigned u; } r;
  r.h[0] = (__bf16)a;
  r.h[1] = (__bf16)b;
  return r.u;
}
__device__ __forceinline__ u16 bf16_bits(float a) {
  union { __bf16 h; u16 u; } r;
  r.h = (__bf16)a;
  return r.u;
}

// ---- staging: each thread moves 64B (32 K-elems) of one tile row per stage --
struct SRF { float4 f[8]; };   // fp32 source: global -> regs -> (convert) -> LDS
struct SRH {};                 // bf16 source: async DMA global -> LDS
template <typename T> struct SRof { using type = SRH; };
template <> struct SRof<float> { using type = SRF; };

__device__ __forceinline__ void stage_issue(const float* __restrict__ g, u16*, SRF& r) {
#pragma unroll
  for (int i = 0; i < 8; ++i) r.f[i] = ((const float4*)g)[i];
}
__device__ __forceinline__ void stage_commit(u16* s, SRF& r) {
#pragma unroll
  for (int i = 0; i < 8; ++i) {
    uint2 q;
    q.x = pack2_bf16(r.f[i].x, r.f[i].y);
    q.y = pack2_bf16(r.f[i].z, r.f[i].w);
    ((uint2*)s)[i] = q;
  }
}
// bf16 source: 4x async b128 loads straight into LDS (tracked by ASYNCcnt).
// LDS aperture maps flat addr[31:0] to the wave-relative LDS offset.
__device__ __forceinline__ void stage_issue(const u16* __restrict__ g, u16* s, SRH&) {
  unsigned lds = (unsigned)(unsigned long long)s;
  asm volatile("global_load_async_to_lds_b128 %0, %1, off offset:0"  :: "v"(lds), "v"(g) : "memory");
  asm volatile("global_load_async_to_lds_b128 %0, %1, off offset:16" :: "v"(lds), "v"(g) : "memory");
  asm volatile("global_load_async_to_lds_b128 %0, %1, off offset:32" :: "v"(lds), "v"(g) : "memory");
  asm volatile("global_load_async_to_lds_b128 %0, %1, off offset:48" :: "v"(lds), "v"(g) : "memory");
}
__device__ __forceinline__ void stage_commit(u16*, SRH&) {}

union Frag { uint4 u[2]; v16bf v; };

// A frag (16x32 bf16): lane L -> row L%16; K base=(L/16)*8, chunks [base,base+8) and [base+16,base+24).
__device__ __forceinline__ v16bf load_frag_A(const u16* sm, int lane) {
  const u16* p = sm + (lane & 15) * LDS_STRIDE + ((lane >> 4) << 3);
  Frag f;
  f.u[0] = *(const uint4*)p;
  f.u[1] = *(const uint4*)(p + 16);
  return f.v;
}
// B frag (32x16 bf16), LDS holds B as [n][k]: lane L -> col L%16; K base=(L/16)*8, chunks [base,base+8) and [base+16,base+24).
__device__ __forceinline__ v16bf load_frag_B(const u16* sm, int lane) {
  const u16* p = sm + (lane & 15) * LDS_STRIDE + ((lane >> 4) << 3);
  Frag f;
  f.u[0] = *(const uint4*)p;
  f.u[1] = *(const uint4*)(p + 16);
  return f.v;
}

// C = A[M,K] * B[N,K]^T (both K-contiguous). 8 waves, each a 32(M)x64(N) sub-tile.
template <typename TA, typename TB, int MODE>
__global__ __launch_bounds__(NTHREADS)
void gemm_wmma(const TA* __restrict__ Araw, const TB* __restrict__ Braw,
               void* __restrict__ Outraw, const unsigned char* __restrict__ Mask,
               int M, int N, int Kdim,
               size_t strideAb, size_t strideBb, size_t strideOb, size_t strideMb,
               float scale) {
  __shared__ u16 sA[2][BM * LDS_STRIDE];
  __shared__ u16 sB[2][BN * LDS_STRIDE];

  constexpr bool asyncAny = (sizeof(TA) == 2) || (sizeof(TB) == 2);

  const int b   = blockIdx.z;
  const int t   = threadIdx.x;
  const int row = t >> 1;          // 0..127: tile row this thread stages
  const int kh  = (t & 1) * 32;    // which 32-wide K half of the stage

  const TA* Abase = Araw + strideAb * b + (size_t)(blockIdx.x * BM + row) * Kdim + kh;
  const TB* Bbase = Braw + strideBb * b + (size_t)(blockIdx.y * BN + row) * Kdim + kh;
  u16* sArow[2] = { &sA[0][row * LDS_STRIDE + kh], &sA[1][row * LDS_STRIDE + kh] };
  u16* sBrow[2] = { &sB[0][row * LDS_STRIDE + kh], &sB[1][row * LDS_STRIDE + kh] };

  const int wid  = t >> 5;
  const int lane = t & 31;
  const int wm   = wid & 3;    // M-offset wm*32
  const int wn   = wid >> 2;   // N-offset wn*64

  const v8f vzero = {0.f, 0.f, 0.f, 0.f, 0.f, 0.f, 0.f, 0.f};
  v8f acc[2][4];
#pragma unroll
  for (int i = 0; i < 2; ++i)
#pragma unroll
    for (int j = 0; j < 4; ++j) acc[i][j] = vzero;

  typename SRof<TA>::type ra;
  typename SRof<TB>::type rb;

  // Prologue: stage k0=0 into buffer 0.
  stage_issue(Abase, sArow[0], ra);
  stage_issue(Bbase, sBrow[0], rb);
  stage_commit(sArow[0], ra);
  stage_commit(sBrow[0], rb);
  if constexpr (asyncAny) asm volatile("s_wait_asynccnt 0x0" ::: "memory");
  __syncthreads();

  int p = 0;
  for (int k0 = 0; k0 < Kdim; k0 += BK) {
    const bool notLast = (k0 + BK) < Kdim;
    if (notLast) {  // issue next stage into the other buffer before computing
      stage_issue(Abase + k0 + BK, sArow[p ^ 1], ra);
      stage_issue(Bbase + k0 + BK, sBrow[p ^ 1], rb);
    }

#pragma unroll
    for (int ks = 0; ks < 2; ++ks) {       // two 16x16x32 K-slices per stage
      v16bf afr[2], bfr[4];
#pragma unroll
      for (int i = 0; i < 2; ++i)
        afr[i] = load_frag_A(&sA[p][(wm * 32 + i * 16) * LDS_STRIDE + ks * 32], lane);
#pragma unroll
      for (int j = 0; j < 4; ++j)
        bfr[j] = load_frag_B(&sB[p][(wn * 64 + j * 16) * LDS_STRIDE + ks * 32], lane);
#pragma unroll
      for (int i = 0; i < 2; ++i)
#pragma unroll
        for (int j = 0; j < 4; ++j)
          acc[i][j] = __builtin_amdgcn_wmma_f32_16x16x32_bf16(
              false, afr[i], false, bfr[j], (short)0, acc[i][j], false, false);
    }

    if (notLast) {
      stage_commit(sArow[p ^ 1], ra);
      stage_commit(sBrow[p ^ 1], rb);
      if constexpr (asyncAny) asm volatile("s_wait_asynccnt 0x0" ::: "memory");
    }
    __syncthreads();
    p ^= 1;
  }

  // Epilogue. f32 C layout: element (m = vgpr + 8*(lane/16), n = lane%16).
  const int nIn   = lane & 15;
  const int mHalf = (lane >> 4) * 8;
#pragma unroll
  for (int i = 0; i < 2; ++i) {
#pragma unroll
    for (int j = 0; j < 4; ++j) {
      const int mBase = blockIdx.x * BM + wm * 32 + i * 16 + mHalf;
      const int n     = blockIdx.y * BN + wn * 64 + j * 16 + nIn;
      if constexpr (MODE == OUT_BF16_T) {
        // transposed store [N][M]: 8 consecutive m per lane -> one b128 store
        u16* out = (u16*)Outraw + strideOb * b;
        uint4 pk;
        pk.x = pack2_bf16(acc[i][j][0], acc[i][j][1]);
        pk.y = pack2_bf16(acc[i][j][2], acc[i][j][3]);
        pk.z = pack2_bf16(acc[i][j][4], acc[i][j][5]);
        pk.w = pack2_bf16(acc[i][j][6], acc[i][j][7]);
        *(uint4*)&out[(size_t)n * M + mBase] = pk;
      } else {
#pragma unroll
        for (int v = 0; v < 8; ++v) {
          const int m = mBase + v;
          float val = acc[i][j][v];
          if constexpr (MODE == OUT_BF16_N) {
            u16* out = (u16*)Outraw + strideOb * b;
            out[(size_t)m * N + n] = bf16_bits(val);
          } else if constexpr (MODE == OUT_F32_SCORES) {
            float* out = (float*)Outraw + strideOb * b;
            val *= scale;
            if (Mask[strideMb * b + (size_t)m * N + n]) val = -1e9f;
            out[(size_t)m * N + n] = val;
          } else {
            float* out = (float*)Outraw + strideOb * b;
            out[(size_t)m * N + n] = val;
          }
        }
      }
    }
  }
}

// In-place row softmax over 2048 f32 columns; one 256-thread block per row.
__global__ __launch_bounds__(256)
void softmax_inplace(float* __restrict__ scores, int cols /* == 2048 */) {
  float* r = scores + (size_t)blockIdx.x * cols;
  const int t = threadIdx.x;
  float v[8];
#pragma unroll
  for (int i = 0; i < 8; ++i) v[i] = r[t + i * 256];

  __shared__ float red[256];
  float m = v[0];
#pragma unroll
  for (int i = 1; i < 8; ++i) m = fmaxf(m, v[i]);
  red[t] = m;
  __syncthreads();
  for (int s = 128; s > 0; s >>= 1) {
    if (t < s) red[t] = fmaxf(red[t], red[t + s]);
    __syncthreads();
  }
  m = red[0];
  __syncthreads();

  float sum = 0.f;
#pragma unroll
  for (int i = 0; i < 8; ++i) { v[i] = __expf(v[i] - m); sum += v[i]; }
  red[t] = sum;
  __syncthreads();
  for (int s = 128; s > 0; s >>= 1) {
    if (t < s) red[t] += red[t + s];
    __syncthreads();
  }
  const float inv = 1.0f / red[0];
#pragma unroll
  for (int i = 0; i < 8; ++i) r[t + i * 256] = v[i] * inv;
}

extern "C" void kernel_launch(void* const* d_in, const int* in_sizes, int n_in,
                              void* d_out, int out_size, void* d_ws, size_t ws_size,
                              hipStream_t stream) {
  // setup_inputs order: x, enc_q, enc_k, i_mask(bool -> 1 byte), Wq, Wk, Wv
  const float* x      = (const float*)d_in[0];
  const float* enc_q  = (const float*)d_in[1];
  const float* enc_k  = (const float*)d_in[2];
  const unsigned char* i_mask = (const unsigned char*)d_in[3];
  const float* Wq = (const float*)d_in[4];
  const float* Wk = (const float*)d_in[5];
  const float* Wv = (const float*)d_in[6];
  float* out = (float*)d_out;

  const int B = 4, SQ = 2048, SKV = 2048, D = 2048;
  const float scale = 1.0f / sqrtf((float)D);

  // Workspace: Q(bf16) 32MiB | K(bf16) 32MiB | Vt(bf16) 32MiB | scores(f32) 64MiB
  u16*  qb  = (u16*)d_ws;
  u16*  kb  = qb  + (size_t)B * SQ  * D;
  u16*  vtb = kb  + (size_t)B * SKV * D;
  float* sc = (float*)(vtb + (size_t)B * SKV * D);

  const dim3 blk(NTHREADS);

  // Q = enc_q @ Wq^T   (bf16, [b][s][e])
  gemm_wmma<float, float, OUT_BF16_N><<<dim3(SQ / BM, D / BN, B), blk, 0, stream>>>(
      enc_q, Wq, qb, nullptr, SQ, D, D,
      (size_t)SQ * D, 0, (size_t)SQ * D, 0, 1.0f);
  // K = enc_k @ Wk^T   (bf16, [b][s][e])
  gemm_wmma<float, float, OUT_BF16_N><<<dim3(SKV / BM, D / BN, B), blk, 0, stream>>>(
      enc_k, Wk, kb, nullptr, SKV, D, D,
      (size_t)SKV * D, 0, (size_t)SKV * D, 0, 1.0f);
  // Vt = (x @ Wv^T)^T  (bf16, [b][e][kv]) so PV GEMM gets a K-contiguous B-operand
  gemm_wmma<float, float, OUT_BF16_T><<<dim3(SKV / BM, D / BN, B), blk, 0, stream>>>(
      x, Wv, vtb, nullptr, SKV, D, D,
      (size_t)SKV * D, 0, (size_t)D * SKV, 0, 1.0f);
  // scores = Q @ K^T * scale, mask -> -1e9  (f32)  [both operands async-staged]
  gemm_wmma<u16, u16, OUT_F32_SCORES><<<dim3(SQ / BM, SKV / BN, B), blk, 0, stream>>>(
      qb, kb, sc, i_mask, SQ, SKV, D,
      (size_t)SQ * D, (size_t)SKV * D, (size_t)SQ * SKV, (size_t)SQ * SKV, scale);
  // softmax rows in place (f32)
  softmax_inplace<<<dim3(B * SQ), dim3(256), 0, stream>>>(sc, SKV);
  // out = attn @ V  (A: f32 attn converted in staging; B: bf16 Vt async-staged)
  gemm_wmma<float, u16, OUT_F32_PLAIN><<<dim3(SQ / BM, D / BN, B), blk, 0, stream>>>(
      sc, vtb, out, nullptr, SQ, D, SKV,
      (size_t)SQ * SKV, (size_t)D * SKV, (size_t)SQ * D, 0, 1.0f);
}
